// Temporal_Aggregation_Module_42339787604715
// MI455X (gfx1250) — matrix operand order repaired and checked by reading, not compiled, not run on hardware
//
#include <hip/hip_runtime.h>
#include <hip/hip_bf16.h>

// ---------------------------------------------------------------------------
// Problem constants (from reference)
// ---------------------------------------------------------------------------
constexpr int BATCH = 16;
constexpr int CIN   = 512;
constexpr int CH    = 256;
constexpr int HW    = 4096;   // 64*64
constexpr int NREF  = 3;
constexpr int KCAT  = CIN + CH;  // 768
constexpr int NK    = NREF * CH; // 768

// ---------------------------------------------------------------------------
// WMMA plumbing (gfx1250, wave32): v_wmma_f32_16x16x32_f16, f32 accumulate
// ---------------------------------------------------------------------------
typedef __attribute__((ext_vector_type(16))) _Float16     v16h;
typedef __attribute__((ext_vector_type(8)))  float        v8f;
typedef __attribute__((ext_vector_type(8)))  unsigned int v8u;

__device__ __forceinline__ unsigned int pack2(float a, float b) {
    union { _Float16 h[2]; unsigned int u; } x;
    x.h[0] = (_Float16)a;
    x.h[1] = (_Float16)b;
    return x.u;
}

// LDS tiles are stored as packed half2 (u32), K-pairs along the minor axis,
// row stride padded to 17 u32 to break bank conflicts.
constexpr int LDSP = 17;

// A fragment: 16x32 (MxK) f16.  Lane L: M = L&15.
// lanes 0-15: VGPR j<4 -> K=2j,2j+1 ; VGPR j>=4 -> K=16+2(j-4),...
// lanes 16-31: same but K base +8.  (CDNA5 ISA 7.12.2)
__device__ __forceinline__ v16h frag_a(const unsigned int* As, int mbase, int lane) {
    const unsigned int* r = As + (mbase + (lane & 15)) * LDSP;
    const int kb = (lane & 16) ? 4 : 0;   // u32-column base (K halves /2)
    v8u u;
#pragma unroll
    for (int j = 0; j < 4; ++j) u[j]     = r[kb + j];
#pragma unroll
    for (int j = 0; j < 4; ++j) u[4 + j] = r[8 + kb + j];
    return __builtin_bit_cast(v16h, u);
}

// B fragment: 32x16 (KxN) f16.  Lane L: N = L&15.
// same K order as frag_a: lanes 0-15 K=0..7,16..23; lanes 16-31 K=8..15,24..31.
__device__ __forceinline__ v16h frag_b(const unsigned int* Bs, int nbase, int lane) {
    const unsigned int* r = Bs + (nbase + (lane & 15)) * LDSP;
    const int kb = (lane & 16) ? 4 : 0;
    v8u u;
#pragma unroll
    for (int j = 0; j < 4; ++j) u[j]     = r[kb + j];
#pragma unroll
    for (int j = 0; j < 4; ++j) u[4 + j] = r[8 + kb + j];
    return __builtin_bit_cast(v16h, u);
}

__device__ __forceinline__ v8f wmma_f16(v16h a, v16h b, v8f c) {
    // 8 args: (neg_a, A, neg_b, B, c_mod, C, reuse_a, reuse_b)
    return __builtin_amdgcn_wmma_f32_16x16x32_f16(false, a, false, b, (short)0, c,
                                                  false, false);
}

// ---------------------------------------------------------------------------
// Shared 256x64 macro-tile GEMM core.  256 threads = 8 waves; wave w owns the
// 32x64 strip starting at M=32*w (2 A-fragments x 4 B-fragments = 8 WMMA tiles,
// K-step 32).  All waves share the B tile through LDS, so the B operand is
// streamed from HBM exactly once per 256 output rows — this is what makes the
// conv-K GEMM read each 402 MB refs tensor once and halves final-conv traffic.
// ALoad(r,c,kk): packed half2 of A[m0+r][kk+2c .. +1]     (r in 0..255, c in 0..15)
// BLoad(r,c,kk): packed half2 of B[kk+2c .. +1][n0+r]     (r in 0..63,  c in 0..15)
// Epi(m,n,v): consume C[m][n] (m,n relative to the 256x64 tile)
// ---------------------------------------------------------------------------
template <class ALoad, class BLoad, class Epi>
__device__ __forceinline__ void wmma_gemm_256x64(int Ktot, ALoad aload, BLoad bload, Epi epi) {
    __shared__ unsigned int As[256 * LDSP];  // 17.0 KB
    __shared__ unsigned int Bs[64 * LDSP];   //  4.3 KB

    const int tid  = threadIdx.x;
    const int lane = tid & 31;
    const int wv   = tid >> 5;
    const int wm   = wv * 32;   // wave M origin in tile

    v8f acc[2][4] = {};

    for (int kk = 0; kk < Ktot; kk += 32) {
        // A tile 256x16 u32: thread t fills row t (128 contiguous bytes of A)
#pragma unroll
        for (int j = 0; j < 16; ++j) As[tid * LDSP + j] = aload(tid, j, kk);
        // B tile 64x16 u32
#pragma unroll
        for (int j = 0; j < 4; ++j) {
            int e = tid * 4 + j;
            Bs[(e >> 4) * LDSP + (e & 15)] = bload(e >> 4, e & 15, kk);
        }
        __syncthreads();

        v16h a0 = frag_a(As, wm,      lane);
        v16h a1 = frag_a(As, wm + 16, lane);
        v16h b0 = frag_b(Bs, 0,  lane);
        v16h b1 = frag_b(Bs, 16, lane);
        v16h b2 = frag_b(Bs, 32, lane);
        v16h b3 = frag_b(Bs, 48, lane);

        acc[0][0] = wmma_f16(a0, b0, acc[0][0]);
        acc[0][1] = wmma_f16(a0, b1, acc[0][1]);
        acc[0][2] = wmma_f16(a0, b2, acc[0][2]);
        acc[0][3] = wmma_f16(a0, b3, acc[0][3]);
        acc[1][0] = wmma_f16(a1, b0, acc[1][0]);
        acc[1][1] = wmma_f16(a1, b1, acc[1][1]);
        acc[1][2] = wmma_f16(a1, b2, acc[1][2]);
        acc[1][3] = wmma_f16(a1, b3, acc[1][3]);
        __syncthreads();
    }

    // C/D 16x16 f32 layout: lane N = L&15; VGPR r -> M = r + (L>=16 ? 8 : 0)
    const int nlo = lane & 15;
    const int mhi = (lane & 16) ? 8 : 0;
#pragma unroll
    for (int i = 0; i < 2; ++i)
#pragma unroll
        for (int j = 0; j < 4; ++j) {
            int mb = wm + i * 16 + mhi;
            int nc = j * 16 + nlo;
#pragma unroll
            for (int r = 0; r < 8; ++r) epi(mb + r, nc, acc[i][j][r]);
        }
}

// ---------------------------------------------------------------------------
// GEMM kernel instantiations
// ---------------------------------------------------------------------------

// kf[b,o,n] = sum_c W[o,c]*X[b,c,n] + bias[o]   (M=CH, N=HW, K=CIN)
__global__ void k_conv_gemm(const float* __restrict__ X, const float* __restrict__ W,
                            const float* __restrict__ bias, float* __restrict__ Y) {
    const int b  = blockIdx.z;
    const int n0 = blockIdx.x * 64;
    const float* Xb = X + (size_t)b * CIN * HW;
    float*       Yb = Y + (size_t)b * CH * HW;
    wmma_gemm_256x64(CIN,
        [&](int r, int c, int kk) {
            const float* p = W + (size_t)r * CIN + kk + c * 2;
            return pack2(p[0], p[1]);
        },
        [&](int r, int c, int kk) {
            const float* p = Xb + (size_t)(kk + c * 2) * HW + n0 + r;
            return pack2(p[0], p[HW]);
        },
        [&](int m, int n, float v) {
            Yb[(size_t)m * HW + n0 + n] = v + bias[m];
        });
}

// P[b, m, c] = sum_n softmax_row(T[b,m,:])[n] * T[b,c,n]   (M=CH, N=CH, K=HW)
__global__ void k_ocr_gemm(const float* __restrict__ T, const float* __restrict__ rmax,
                           const float* __restrict__ rrcp, float* __restrict__ P,
                           size_t bstride) {
    const int b  = blockIdx.z;
    const int n0 = blockIdx.x * 64;
    const float* Tb = T + (size_t)b * CH * HW;
    float*       Pb = P + (size_t)b * bstride;
    wmma_gemm_256x64(HW,
        [&](int r, int c, int kk) {
            int rowg = b * CH + r;
            float mx = rmax[rowg], rc = rrcp[rowg];
            const float* p = Tb + (size_t)r * HW + kk + c * 2;
            return pack2(__expf(p[0] - mx) * rc, __expf(p[1] - mx) * rc);
        },
        [&](int r, int c, int kk) {
            const float* p = Tb + (size_t)(n0 + r) * HW + kk + c * 2;
            return pack2(p[0], p[1]);
        },
        [&](int m, int n, float v) {
            Pb[(size_t)m * CH + n0 + n] = v;
        });
}

// sim[b,q,k] = (1/16) sum_c qpb[b,q,c]*kpbs[b,k,c]   (M=CH, N=NK, K=CH)
__global__ void k_sim_gemm(const float* __restrict__ Qp, const float* __restrict__ Kp,
                           float* __restrict__ S) {
    const int b  = blockIdx.z;
    const int n0 = blockIdx.x * 64;
    const float* Qb = Qp + (size_t)b * CH * CH;
    const float* Kb = Kp + (size_t)b * NK * CH;
    float*       Sb = S  + (size_t)b * CH * NK;
    wmma_gemm_256x64(CH,
        [&](int r, int c, int kk) {
            const float* p = Qb + (size_t)r * CH + kk + c * 2;
            return pack2(p[0], p[1]);
        },
        [&](int r, int c, int kk) {
            const float* p = Kb + (size_t)(n0 + r) * CH + kk + c * 2;
            return pack2(p[0], p[1]);
        },
        [&](int m, int n, float v) {
            Sb[(size_t)m * NK + n0 + n] = v * 0.0625f;  // 1/sqrt(256)
        });
}

// value[b,q,c] = sum_k sim[b,q,k]*kpbs[b,k,c]   (M=CH, N=CH, K=NK)
__global__ void k_value_gemm(const float* __restrict__ S, const float* __restrict__ Kp,
                             float* __restrict__ V) {
    const int b  = blockIdx.z;
    const int n0 = blockIdx.x * 64;
    const float* Sb = S  + (size_t)b * CH * NK;
    const float* Kb = Kp + (size_t)b * NK * CH;
    float*       Vb = V  + (size_t)b * CH * CH;
    wmma_gemm_256x64(NK,
        [&](int r, int c, int kk) {
            const float* p = Sb + (size_t)r * NK + kk + c * 2;
            return pack2(p[0], p[1]);
        },
        [&](int r, int c, int kk) {
            const float* p = Kb + (size_t)(kk + c * 2) * CH + n0 + r;
            return pack2(p[0], p[CH]);
        },
        [&](int m, int n, float v) {
            Vb[(size_t)m * CH + n0 + n] = v;
        });
}

// cm[b,p,n] = sum_c (value[b,p,c]*vnorm[b,p]) * (Q[b,c,n]*pixn[b,n])  (M=CH,N=HW,K=CH)
__global__ void k_cm_gemm(const float* __restrict__ V, const float* __restrict__ vnorm,
                          const float* __restrict__ Q, const float* __restrict__ pixn,
                          float* __restrict__ CM) {
    const int b  = blockIdx.z;
    const int n0 = blockIdx.x * 64;
    const float* Vb = V  + (size_t)b * CH * CH;
    const float* Qb = Q  + (size_t)b * CH * HW;
    const float* pn = pixn + (size_t)b * HW;
    float*      CMb = CM + (size_t)b * CH * HW;
    wmma_gemm_256x64(CH,
        [&](int r, int c, int kk) {
            float sc = vnorm[b * CH + r];
            const float* p = Vb + (size_t)r * CH + kk + c * 2;
            return pack2(p[0] * sc, p[1] * sc);
        },
        [&](int r, int c, int kk) {
            int n = n0 + r;
            float sc = pn[n];
            const float* p = Qb + (size_t)(kk + c * 2) * HW + n;
            return pack2(p[0] * sc, p[HW] * sc);
        },
        [&](int m, int n, float v) {
            CMb[(size_t)m * HW + n0 + n] = v;
        });
}

// y[b,o,n] = sum_{k<768} fw[o,k]*cat[k,n] + fb[o]; cat = [x(512) ; cm(256)]
__global__ void k_final_gemm(const float* __restrict__ X, const float* __restrict__ CM,
                             const float* __restrict__ Fw, const float* __restrict__ Fb,
                             float* __restrict__ Y) {
    const int b  = blockIdx.z;
    const int m0 = blockIdx.y * 256;
    const int n0 = blockIdx.x * 64;
    const float* Xb  = X  + (size_t)b * CIN * HW;
    const float* CMb = CM + (size_t)b * CH * HW;
    float*       Yb  = Y  + (size_t)b * CIN * HW;
    wmma_gemm_256x64(KCAT,
        [&](int r, int c, int kk) {
            const float* p = Fw + (size_t)(m0 + r) * KCAT + kk + c * 2;
            return pack2(p[0], p[1]);
        },
        [&](int r, int c, int kk) {
            int k = kk + c * 2;   // pair never straddles the 512 boundary
            const float* p = (k < CIN) ? (Xb + (size_t)k * HW + n0 + r)
                                       : (CMb + (size_t)(k - CIN) * HW + n0 + r);
            return pack2(p[0], p[HW]);
        },
        [&](int m, int n, float v) {
            Yb[(size_t)(m0 + m) * HW + n0 + n] = v + Fb[m0 + m];
        });
}

// ---------------------------------------------------------------------------
// Reduction / elementwise kernels
// ---------------------------------------------------------------------------

// Per-row (len HW) max and reciprocal exp-sum for softmax.  grid = B*CH rows.
__global__ void k_rowstats(const float* __restrict__ T, float* __restrict__ rmax,
                           float* __restrict__ rrcp) {
    const int row = blockIdx.x, t = threadIdx.x;
    const float* p = T + (size_t)row * HW;
    float v[16];
    float mx = -3.4e38f;
#pragma unroll
    for (int j = 0; j < 16; ++j) { v[j] = p[t + j * 256]; mx = fmaxf(mx, v[j]); }
    __shared__ float red[256];
    red[t] = mx; __syncthreads();
    for (int s = 128; s > 0; s >>= 1) { if (t < s) red[t] = fmaxf(red[t], red[t + s]); __syncthreads(); }
    mx = red[0]; __syncthreads();
    float sm = 0.f;
#pragma unroll
    for (int j = 0; j < 16; ++j) sm += __expf(v[j] - mx);
    red[t] = sm; __syncthreads();
    for (int s = 128; s > 0; s >>= 1) { if (t < s) red[t] += red[t + s]; __syncthreads(); }
    if (t == 0) { rmax[row] = mx; rrcp[row] = 1.0f / red[0]; }
}

// In-place softmax over rows of length NK=768.  grid = B*CH rows, 256 threads.
__global__ void k_softmax768(float* __restrict__ S) {
    const int row = blockIdx.x, t = threadIdx.x;
    float* p = S + (size_t)row * NK;
    float v0 = p[t], v1 = p[t + 256], v2 = p[t + 512];
    __shared__ float red[256];
    red[t] = fmaxf(v0, fmaxf(v1, v2)); __syncthreads();
    for (int s = 128; s > 0; s >>= 1) { if (t < s) red[t] = fmaxf(red[t], red[t + s]); __syncthreads(); }
    float mx = red[0]; __syncthreads();
    v0 = __expf(v0 - mx); v1 = __expf(v1 - mx); v2 = __expf(v2 - mx);
    red[t] = v0 + v1 + v2; __syncthreads();
    for (int s = 128; s > 0; s >>= 1) { if (t < s) red[t] += red[t + s]; __syncthreads(); }
    float rc = 1.0f / red[0];
    p[t] = v0 * rc; p[t + 256] = v1 * rc; p[t + 512] = v2 * rc;
}

// Inverse L2 norm of each 256-wide row of value.  grid = B*CH, 256 threads.
__global__ void k_vnorm(const float* __restrict__ V, float* __restrict__ out) {
    const int row = blockIdx.x, t = threadIdx.x;
    float v = V[(size_t)row * CH + t];
    __shared__ float red[256];
    red[t] = v * v; __syncthreads();
    for (int s = 128; s > 0; s >>= 1) { if (t < s) red[t] += red[t + s]; __syncthreads(); }
    if (t == 0) out[row] = rsqrtf(red[0]);
}

// Inverse L2 norm over channels for each pixel of query.  grid covers B*HW.
__global__ void k_pixnorm(const float* __restrict__ Q, float* __restrict__ out) {
    const int idx = blockIdx.x * 256 + threadIdx.x;  // b*HW + n
    const int b = idx >> 12, n = idx & (HW - 1);
    const float* p = Q + (size_t)b * CH * HW + n;
    float s = 0.f;
    for (int c = 0; c < CH; ++c) { float v = p[(size_t)c * HW]; s += v * v; }
    out[idx] = rsqrtf(s);
}

// BN batch stats per channel o over (B, HW).  grid = CIN, 256 threads.
__global__ void k_bnstats(const float* __restrict__ Y, float* __restrict__ mean,
                          float* __restrict__ istd) {
    const int o = blockIdx.x, t = threadIdx.x;
    float s = 0.f, sq = 0.f;
    for (int b = 0; b < BATCH; ++b) {
        const float* p = Y + ((size_t)b * CIN + o) * HW;
        for (int n = t; n < HW; n += 256) { float v = p[n]; s += v; sq += v * v; }
    }
    __shared__ float r0[256], r1[256];
    r0[t] = s; r1[t] = sq; __syncthreads();
    for (int st = 128; st > 0; st >>= 1) {
        if (t < st) { r0[t] += r0[t + st]; r1[t] += r1[t + st]; }
        __syncthreads();
    }
    if (t == 0) {
        const float inv_n = 1.0f / (float)(BATCH * HW);
        float m = r0[0] * inv_n;
        float var = r1[0] * inv_n - m * m;
        mean[o] = m;
        istd[o] = rsqrtf(var + 1e-5f);
    }
}

// Normalize + affine + ReLU, in place on d_out.
__global__ void k_bnapply(float* __restrict__ Y, const float* __restrict__ mean,
                          const float* __restrict__ istd, const float* __restrict__ g,
                          const float* __restrict__ bb) {
    const size_t idx = (size_t)blockIdx.x * 256 + threadIdx.x;
    const int o = (int)((idx >> 12) & (CIN - 1));
    float v = (Y[idx] - mean[o]) * istd[o] * g[o] + bb[o];
    Y[idx] = fmaxf(v, 0.f);
}

// ---------------------------------------------------------------------------
// Host launcher
// ---------------------------------------------------------------------------
extern "C" void kernel_launch(void* const* d_in, const int* in_sizes, int n_in,
                              void* d_out, int out_size, void* d_ws, size_t ws_size,
                              hipStream_t stream) {
    const float* x     = (const float*)d_in[0];   // [B,CIN,H,W]
    const float* query = (const float*)d_in[1];   // [B,CH,H,W]
    const float* refs  = (const float*)d_in[2];   // [N,B,CIN,H,W]
    const float* K_w   = (const float*)d_in[3];   // [CH,CIN]
    const float* K_b   = (const float*)d_in[4];   // [CH]
    // d_in[5], d_in[6] = V_w, V_b: dead in the reference (results discarded)
    const float* f_w   = (const float*)d_in[7];   // [CIN, CIN+CH]
    const float* f_b   = (const float*)d_in[8];   // [CIN]
    const float* bn_g  = (const float*)d_in[9];   // [CIN]
    const float* bn_b  = (const float*)d_in[10];  // [CIN]
    float* out = (float*)d_out;                   // [B,CIN,H,W]

    // Workspace layout (floats). ~101 MB total; cm reuses the kf region.
    float* ws    = (float*)d_ws;
    float* kf    = ws;                                        // B*CH*HW   = 16,777,216
    float* k_pbs = kf    + (size_t)BATCH * CH * HW;           // B*NK*CH   =  3,145,728
    float* q_pb  = k_pbs + (size_t)BATCH * NK * CH;           // B*CH*CH   =  1,048,576
    float* sim   = q_pb  + (size_t)BATCH * CH * CH;           // B*CH*NK   =  3,145,728
    float* value = sim   + (size_t)BATCH * CH * NK;           // B*CH*CH   =  1,048,576
    float* rmax  = value + (size_t)BATCH * CH * CH;           // B*CH      =      4,096
    float* rrcp  = rmax  + BATCH * CH;                        // B*CH
    float* vnorm = rrcp  + BATCH * CH;                        // B*CH
    float* pixn  = vnorm + BATCH * CH;                        // B*HW      =     65,536
    float* bnm   = pixn  + BATCH * HW;                        // CIN
    float* bni   = bnm   + CIN;                               // CIN
    float* cm    = kf;   // kf no longer needed once k_pbs is built

    const dim3 blk(256);

    // Per-reference: conv-K GEMM -> softmax row stats -> OCR GEMM into k_pbs slice
    for (int i = 0; i < NREF; ++i) {
        const float* ref_i = refs + (size_t)i * BATCH * CIN * HW;
        k_conv_gemm<<<dim3(HW / 64, 1, BATCH), blk, 0, stream>>>(ref_i, K_w, K_b, kf);
        k_rowstats<<<dim3(BATCH * CH), blk, 0, stream>>>(kf, rmax, rrcp);
        k_ocr_gemm<<<dim3(CH / 64, 1, BATCH), blk, 0, stream>>>(
            kf, rmax, rrcp, k_pbs + (size_t)i * CH * CH, (size_t)NK * CH);
    }

    // Query OCR (operates directly on query)
    k_rowstats<<<dim3(BATCH * CH), blk, 0, stream>>>(query, rmax, rrcp);
    k_ocr_gemm<<<dim3(CH / 64, 1, BATCH), blk, 0, stream>>>(
        query, rmax, rrcp, q_pb, (size_t)CH * CH);

    // Attention: sim -> softmax -> value
    k_sim_gemm<<<dim3(NK / 64, 1, BATCH), blk, 0, stream>>>(q_pb, k_pbs, sim);
    k_softmax768<<<dim3(BATCH * CH), blk, 0, stream>>>(sim);
    k_value_gemm<<<dim3(CH / 64, 1, BATCH), blk, 0, stream>>>(sim, k_pbs, value);

    // Correlation map
    k_vnorm<<<dim3(BATCH * CH), blk, 0, stream>>>(value, vnorm);
    k_pixnorm<<<dim3(BATCH * HW / 256), blk, 0, stream>>>(query, pixn);
    k_cm_gemm<<<dim3(HW / 64, 1, BATCH), blk, 0, stream>>>(value, vnorm, query, pixn, cm);

    // Final 1x1 conv over [x ; cm], then BN (batch stats) + ReLU in place
    k_final_gemm<<<dim3(HW / 64, CIN / 256, BATCH), blk, 0, stream>>>(x, cm, f_w, f_b, out);
    k_bnstats<<<dim3(CIN), blk, 0, stream>>>(out, bnm, bni);
    k_bnapply<<<dim3((unsigned)((size_t)BATCH * CIN * HW / 256)), blk, 0, stream>>>(
        out, bnm, bni, bn_g, bn_b);
}
